// RFFR_13975823582014
// MI455X (gfx1250) — compile-verified, matrix operand order checked by reading, not run on hardware
//
#include <hip/hip_runtime.h>

typedef _Float16 v8h  __attribute__((ext_vector_type(8)));
typedef _Float16 v16h __attribute__((ext_vector_type(16)));
typedef float    v8f  __attribute__((ext_vector_type(8)));

#if __has_builtin(__builtin_amdgcn_global_load_async_to_lds_b128) && \
    __has_builtin(__builtin_amdgcn_s_wait_asynccnt)
#define HAS_ASYNC 1
// exact parameter types of the builtin: GCC-style int4 vectors in AS(1)/AS(3)
typedef int v4i __attribute__((vector_size(16)));
typedef __attribute__((address_space(1))) v4i v4i_g;
typedef __attribute__((address_space(3))) v4i v4i_l;
#else
#define HAS_ASYNC 0
#endif

#define B_  16
#define F_  32
#define N_  64
#define D_  512
#define CK      128    // K-chunk in halves staged per step
#define CSTRIDE 136    // J-chunk LDS row stride (halves): 68 dwords, 68%64=4 -> bank spread
#define ISTRIDE 520    // resident-I LDS row stride (halves): 260 dwords, 260%64=4

// ---- dynamic-LDS carve (bytes) ----
#define OFF_I    0
#define SZ_I     (N_ * ISTRIDE * 2)          // 66560: tn_i resident, full K
#define OFF_J0   (OFF_I + SZ_I)
#define SZ_J     (N_ * CSTRIDE * 2)          // 17408: one J K-chunk
#define OFF_J1   (OFF_J0 + SZ_J)
#define OFF_DEN  (OFF_J1 + SZ_J)             // 101376
#define OFF_ACC  (OFF_DEN + N_ * 4)
#define OFF_MIN  (OFF_ACC + N_ * 4)
#define OFF_MAX  (OFF_MIN + 32)
#define LDS_BYTES (OFF_MAX + 32)             // 101952 -> 3 workgroups / WGP (320 KB)

union Frag { v16h v; v8h h[2]; };

// 16-byte global->LDS copy: async DMA path on CDNA5, sync fallback otherwise.
__device__ __forceinline__ void cp16(const _Float16* g, _Float16* l) {
#if HAS_ASYNC
    __builtin_amdgcn_global_load_async_to_lds_b128((v4i_g*)g, (v4i_l*)l, 0, 0);
#else
    *(v8h*)l = *(const v8h*)g;
#endif
}
__device__ __forceinline__ void async_wait0() {
#if HAS_ASYNC
    __builtin_amdgcn_s_wait_asynccnt(0);
#endif
}

// Stage one 64 x 128-half K-chunk of tn_j into an LDS buffer (1024 x 16B, 4/thread).
__device__ __forceinline__ void stage_chunk(const _Float16* __restrict__ tnJ, int kc,
                                            _Float16* buf, int tid) {
    const _Float16* src = tnJ + kc * CK;
    for (int s = tid; s < 1024; s += 256) {
        int rr = s >> 4;
        int c8 = (s & 15) * 8;
        cp16(src + (size_t)rr * D_ + c8, buf + rr * CSTRIDE + c8);
    }
}

// ---------------------------------------------------------------------------
// Phase 1: tn[b,f,n,:] = f16( tgt / ||tgt||_2 )   (one wave per 512-elem row)
// ---------------------------------------------------------------------------
__global__ __launch_bounds__(256) void normalize_f16_kernel(const float* __restrict__ tgt,
                                                            _Float16* __restrict__ tn) {
    int row  = blockIdx.x * 8 + (threadIdx.x >> 5);
    int lane = threadIdx.x & 31;
    const float4* s4 = (const float4*)(tgt + (size_t)row * D_ + lane * 16);
    float tv[16];
    *(float4*)&tv[0]  = s4[0];
    *(float4*)&tv[4]  = s4[1];
    *(float4*)&tv[8]  = s4[2];
    *(float4*)&tv[12] = s4[3];
    float ss = 0.f;
#pragma unroll
    for (int k = 0; k < 16; ++k) ss += tv[k] * tv[k];
#pragma unroll
    for (int m = 16; m >= 1; m >>= 1) ss += __shfl_xor(ss, m, 32);
    float inv = rsqrtf(ss);
    v8h o0, o1;
#pragma unroll
    for (int k = 0; k < 8; ++k) {
        o0[k] = (_Float16)(tv[k] * inv);
        o1[k] = (_Float16)(tv[k + 8] * inv);
    }
    v8h* dst = (v8h*)(tn + (size_t)row * D_ + lane * 16);
    dst[0] = o0;
    dst[1] = o1;
}

// ---------------------------------------------------------------------------
// Phase 2: one workgroup per (b,i).  tn_i resident in LDS; tn_j K-chunks
// double-buffered via async global->LDS DMA, overlapped with WMMA.
// ---------------------------------------------------------------------------
__global__ __launch_bounds__(256) void pairsim_kernel(const float* __restrict__ tgt,
                                                      const _Float16* __restrict__ tn,
                                                      float* __restrict__ out) {
    extern __shared__ char smem[];
    _Float16* ldsI   = (_Float16*)(smem + OFF_I);
    _Float16* jbuf0  = (_Float16*)(smem + OFF_J0);
    _Float16* jbuf1  = (_Float16*)(smem + OFF_J1);
    float*    colden = (float*)(smem + OFF_DEN);
    float*    colacc = (float*)(smem + OFF_ACC);
    float*    wminA  = (float*)(smem + OFF_MIN);
    float*    wmaxA  = (float*)(smem + OFF_MAX);

    const int tid  = threadIdx.x;
    const int w    = tid >> 5;
    const int lane = tid & 31;
    const int r    = lane & 15;
    const int h    = lane >> 4;
    const int bi   = blockIdx.x;          // b*F + i
    const int b    = bi >> 5;
    const int i    = bi & 31;

    const int rn  = w >> 1;               // wave's tile row, tiles (rn,cn0),(rn,cn1)
    const int cn0 = (w & 1) * 2;
    const int cn1 = cn0 + 1;

    const _Float16* tnI = tn + (size_t)bi * (N_ * D_);
    if (tid < N_) colacc[tid] = 0.f;

    // ---- prologue: async-stage resident tn_i (full K) + first J chunk ----
    for (int s = tid; s < 4096; s += 256) {          // 128 rows x 32 segs
        int rr = s >> 5;
        int c8 = (s & 31) * 8;
        cp16(tnI + (size_t)rr * D_ + c8, ldsI + rr * ISTRIDE + c8);
    }
    {
        int j0 = (i == 0) ? 1 : 0;
        stage_chunk(tn + (size_t)(b * F_ + j0) * (N_ * D_), 0, jbuf0, tid);
    }

    int parity = 0;
    for (int jidx = 0; jidx < F_ - 1; ++jidx) {
        const int j  = jidx + (jidx >= i ? 1 : 0);
        const int nn = jidx + 1;
        const int jn = (nn < F_ - 1) ? (nn + (nn >= i ? 1 : 0)) : -1;
        const _Float16* tnJ  = tn + (size_t)(b * F_ + j) * (N_ * D_);
        const _Float16* tnJn = (jn >= 0) ? tn + (size_t)(b * F_ + jn) * (N_ * D_) : nullptr;

        v8f c0 = {};
        v8f c1 = {};
        for (int kc = 0; kc < 4; ++kc) {
            async_wait0();               // my DMA for chunk kc has landed in LDS
            __syncthreads();             // everyone's landed; prev buffer reads done
            if (kc == 0 && tid < N_) colden[tid] = 0.f;

            _Float16* cur = parity ? jbuf1 : jbuf0;
            _Float16* nxt = parity ? jbuf0 : jbuf1;
            if (kc < 3)        stage_chunk(tnJ,  kc + 1, nxt, tid);   // prefetch next chunk
            else if (jn >= 0)  stage_chunk(tnJn, 0,      nxt, tid);   // prefetch next j

#pragma unroll
            for (int ks = 0; ks < 4; ++ks) {
                const int kb = ks * 32;
                Frag a, b0, b1;
                int ao = (rn * 16 + r) * ISTRIDE + kc * CK + kb + 8 * h;
                a.h[0] = *(const v8h*)&ldsI[ao];
                a.h[1] = *(const v8h*)&ldsI[ao + 16];
                int bo0 = (cn0 * 16 + r) * CSTRIDE + kb + 8 * h;
                b0.h[0] = *(const v8h*)&cur[bo0];
                b0.h[1] = *(const v8h*)&cur[bo0 + 16];
                int bo1 = (cn1 * 16 + r) * CSTRIDE + kb + 8 * h;
                b1.h[0] = *(const v8h*)&cur[bo1];
                b1.h[1] = *(const v8h*)&cur[bo1 + 16];
                c0 = __builtin_amdgcn_wmma_f32_16x16x32_f16(false, a.v, false, b0.v,
                                                            (short)0, c0, false, false);
                c1 = __builtin_amdgcn_wmma_f32_16x16x32_f16(false, a.v, false, b1.v,
                                                            (short)0, c1, false, false);
            }
            parity ^= 1;
        }

        // ---- global min/max over the 64x64 S ----
        float mn = c0[0], mx = c0[0];
#pragma unroll
        for (int v = 0; v < 8; ++v) {
            mn = fminf(mn, fminf(c0[v], c1[v]));
            mx = fmaxf(mx, fmaxf(c0[v], c1[v]));
        }
#pragma unroll
        for (int m = 16; m >= 1; m >>= 1) {
            mn = fminf(mn, __shfl_xor(mn, m, 32));
            mx = fmaxf(mx, __shfl_xor(mx, m, 32));
        }
        if (lane == 0) { wminA[w] = mn; wmaxA[w] = mx; }
        __syncthreads();
        mn = wminA[0]; mx = wmaxA[0];
#pragma unroll
        for (int k = 1; k < 8; ++k) { mn = fminf(mn, wminA[k]); mx = fmaxf(mx, wmaxA[k]); }
        const float inv = 1.f / (mx - mn);

        // ---- e = exp(w), per-tile half-column sums (rows 0-7 / 8-15) ----
        float hs0 = 0.f, hs1 = 0.f;
#pragma unroll
        for (int v = 0; v < 8; ++v) {
            float e0 = __expf(1.f - (c0[v] - mn) * inv);
            float e1 = __expf(1.f - (c1[v] - mn) * inv);
            c0[v] = e0; c1[v] = e1;
            hs0 += e0; hs1 += e1;
        }
        float fs0 = hs0 + __shfl_xor(hs0, 16, 32);
        float fs1 = hs1 + __shfl_xor(hs1, 16, 32);
        if (lane < 16) {
            atomicAdd(&colden[cn0 * 16 + r], fs0);
            atomicAdd(&colden[cn1 * 16 + r], fs1);
        }
        __syncthreads();

        // ---- softmax-normalized column sums, accumulate over j ----
        float p0 = hs0 / colden[cn0 * 16 + r];
        float p1 = hs1 / colden[cn1 * 16 + r];
        float q0 = p0 + __shfl_xor(p0, 16, 32);
        float q1 = p1 + __shfl_xor(p1, 16, 32);
        if (lane < 16) {
            atomicAdd(&colacc[cn0 * 16 + r], q0);
            atomicAdd(&colacc[cn1 * 16 + r], q1);
        }
    }
    async_wait0();
    __syncthreads();

    // ---- out[b,i,d] = sum_m colacc[m] * tgt[b,i,m,d]  (256 threads x 2 d) ----
    const float* tg = tgt + (size_t)bi * (N_ * D_);
    float a0 = 0.f, a1 = 0.f;
    for (int m = 0; m < N_; ++m) {
        float cw = colacc[m];
        a0 += cw * tg[(size_t)m * D_ + tid];
        a1 += cw * tg[(size_t)m * D_ + tid + 256];
    }
    out[(size_t)bi * D_ + tid]       = a0;
    out[(size_t)bi * D_ + tid + 256] = a1;
}

// ---------------------------------------------------------------------------
extern "C" void kernel_launch(void* const* d_in, const int* in_sizes, int n_in,
                              void* d_out, int out_size, void* d_ws, size_t ws_size,
                              hipStream_t stream) {
    const float* tgt = (const float*)d_in[0];      // [16,32,64,512] f32
    float* out = (float*)d_out;                    // [16,32,512] f32
    _Float16* tn = (_Float16*)d_ws;                // 32 MB f16 scratch

    normalize_f16_kernel<<<dim3(4096), dim3(256), 0, stream>>>(tgt, tn);
    pairsim_kernel<<<dim3(B_ * F_), dim3(256), LDS_BYTES, stream>>>(tgt, tn, out);
}
